// Network_73847667687561
// MI455X (gfx1250) — matrix operand order repaired and checked by reading, not compiled, not run on hardware
//
#include <hip/hip_runtime.h>
#include <cstdint>
#include <cstddef>

typedef __attribute__((ext_vector_type(16))) _Float16 v16h;
typedef __attribute__((ext_vector_type(8)))  _Float16 v8h;
typedef __attribute__((ext_vector_type(8)))  float    v8f;

#if defined(__has_builtin)
#if __has_builtin(__builtin_amdgcn_global_load_async_to_lds_b32) && __has_builtin(__builtin_amdgcn_s_wait_asynccnt)
#define HAVE_ASYNC_LDS 1
#endif
#endif

// ---------------- geometry constants ----------------
__constant__ int c_H[5]       = {200,100,50,25,13};
__constant__ int c_W[5]       = {320,160,80,40,20};
__constant__ int c_STRIDE[5]  = {4,8,16,32,64};
__constant__ int c_SIZE[5]    = {32,64,128,256,512};
__constant__ int c_PIXOFF[6]  = {0,64000,80000,84000,85000,85260};
// 2D row-segment tiling: tiles-per-row = ceil(W/16)
__constant__ int c_TPR[5]     = {20,10,5,3,2};
__constant__ int c_TILEOFF[6] = {0,4000,5000,5250,5325,5351};
__constant__ int c_CNT[5]     = {1000,1000,1000,1000,780};   // min(PRE_NMS, H*W*3)

#define NPIX_TOTAL   85260
#define NTILES_TOTAL 5351
#define CONV_K       2304      // 256 ic * 9 taps, ordered k = tap*256 + ic
#define CONV_KT      72        // 2304 / 32
#define APITCH       264       // padded ic-pitch of LDS halo (multiple of 8; 528B rows)

// ---------------- workspace layout ----------------
static constexpr size_t alignup(size_t x){ return (x + 255) & ~size_t(255); }
static constexpr size_t OFF_T  = 0;                               // t  f16 [85260][256]
static constexpr size_t SZ_T   = (size_t)NPIX_TOTAL*256*2;
static constexpr size_t OFF_X  = alignup(OFF_T + SZ_T);           // pooled f16 [1000][12544]
static constexpr size_t SZ_X   = 1000ull*12544*2;
static constexpr size_t OFF_H6 = alignup(OFF_X + SZ_X);           // f16 [1000][1024]
static constexpr size_t SZ_H   = 1000ull*1024*2;
static constexpr size_t OFF_H7 = alignup(OFF_H6 + SZ_H);
static constexpr size_t OFF_LG = alignup(OFF_H7 + SZ_H);          // logits f32 [85260*3]
static constexpr size_t SZ_LG  = (size_t)NPIX_TOTAL*3*4;
static constexpr size_t OFF_DL = alignup(OFF_LG + SZ_LG);         // deltas f32 [85260*12]
static constexpr size_t SZ_DL  = (size_t)NPIX_TOTAL*12*4;
static constexpr size_t OFF_SI = alignup(OFF_DL + SZ_DL);         // sel_idx   i32 [5][1000]
static constexpr size_t OFF_SS = alignup(OFF_SI + 5000*4);        // sel_score f32 [5][1000]
static constexpr size_t OFF_SB = alignup(OFF_SS + 5000*4);        // sel_boxes f32 [5][1000][4]
static constexpr size_t OFF_NB = alignup(OFF_SB + 20000*4);       // nms_boxes f32 [5][1000][4]
static constexpr size_t OFF_NS = alignup(OFF_NB + 20000*4);       // nms_scores f32 [5][1000]
static constexpr size_t OFF_PR = alignup(OFF_NS + 5000*4);        // proposals f32 [1000][4]
// prepared (f16) weights
static constexpr size_t OFF_WT = alignup(OFF_PR + 4000*4);        // conv w swizzled f16 [72][16][32][16]
static constexpr size_t SZ_WT  = (size_t)CONV_KT*16*32*16*2;
static constexpr size_t OFF_W6 = alignup(OFF_WT + SZ_WT);         // fc6_w f16 [1024][12544]
static constexpr size_t SZ_W6  = 1024ull*12544*2;
static constexpr size_t OFF_W7 = alignup(OFF_W6 + SZ_W6);         // fc7_w f16 [1024][1024]
static constexpr size_t SZ_W7  = 1024ull*1024*2;
static constexpr size_t OFF_WC = alignup(OFF_W7 + SZ_W7);         // cls_w f16 [81][1024]
static constexpr size_t SZ_WC  = 81ull*1024*2;
static constexpr size_t OFF_WB = alignup(OFF_WC + SZ_WC);         // bbox_w f16 [324][1024]

// ====================================================================
// Prep A: elementwise f32 -> f16 weight conversion
// ====================================================================
__global__ void __launch_bounds__(256)
cvt_f16_kernel(const float* __restrict__ src, _Float16* __restrict__ dst, int n)
{
    int i = blockIdx.x * 256 + threadIdx.x;
    if (i < n) dst[i] = (_Float16)src[i];
}

// ====================================================================
// Prep B: swizzle rpn_conv_w (OIHW) into WMMA B-fragment order, f16.
// K ordering: k = tap*256 + ic  (tap = ky*3+kx).  Storage:
//   wT[(((kt*16)+nT)*32 + lane)*16 + e] = w[n][ic][ky][kx]
//   with k = kt*32 + (lane>>4)*16 + e, n = nT*16 + (lane&15)
// ====================================================================
__global__ void __launch_bounds__(256)
swizzle_convw_kernel(const float* __restrict__ w, _Float16* __restrict__ wT)
{
    int t = blockIdx.x * 256 + threadIdx.x;       // (kt,nT,lane) id
    if (t >= CONV_KT * 16 * 32) return;
    int lane = t & 31;
    int nT   = (t >> 5) & 15;
    int kt   = t >> 9;
    int n = nT * 16 + (lane & 15);
    _Float16* dst = wT + (size_t)t * 16;
    #pragma unroll
    for (int e = 0; e < 16; ++e) {
        int k = kt * 32 + ((lane >> 4) << 4) + e;
        int tap = k >> 8;          // 0..8 -> ky*3+kx
        int ic  = k & 255;
        dst[e] = (_Float16)w[(size_t)n * CONV_K + ic * 9 + tap];
    }
}

// ====================================================================
// Kernel 1: RPN 3x3 conv (implicit GEMM via WMMA f32_16x16x32_f16)
// Block = 256 thr (8 waves). LDS halo laid out [pos(3x18)][ic(pad 264)]
// so each lane's A fragment = two contiguous ds_load_b128.
// Grid: (NTILES_TOTAL, 2)  -> blockIdx.y selects out-channel half.
// ====================================================================
__global__ void __launch_bounds__(256)
rpn_conv3x3_kernel(const float* __restrict__ f0, const float* __restrict__ f1,
                   const float* __restrict__ f2, const float* __restrict__ f3,
                   const float* __restrict__ f4,
                   const _Float16* __restrict__ wT, const float* __restrict__ bias,
                   _Float16* __restrict__ t_out)
{
    __shared__ _Float16 s_a[54 * APITCH];   // 28512 B
    int tileg = blockIdx.x;
    int tid   = threadIdx.x;
    int L = 0;
    while (L < 4 && tileg >= c_TILEOFF[L+1]) ++L;
    int lt  = tileg - c_TILEOFF[L];
    int H = c_H[L], W = c_W[L], tpr = c_TPR[L];
    int npix = H * W;
    int ty = lt / tpr;
    int x0 = (lt - ty * tpr) * 16;
    const float* feat = (L==0)?f0:(L==1)?f1:(L==2)?f2:(L==3)?f3:f4;

    // ---- stage halo: global reads rx-contiguous (coalesced); LDS [pos][ic] ----
    #pragma unroll 1
    for (int i = tid; i < 256 * 54; i += 256) {
        int ic = i / 54;
        int r  = i - ic * 54;          // pos = ry*18 + rx
        int ry = r / 18;
        int rx = r - ry * 18;
        int iy = ty + ry - 1;
        int ix = x0 + rx - 1;
        float v = 0.0f;
        if (iy >= 0 && iy < H && ix >= 0 && ix < W)
            v = feat[(size_t)ic * npix + (size_t)iy * W + ix];
        s_a[r * APITCH + ic] = (_Float16)v;
    }
    __syncthreads();

    // ---- per-wave WMMA over K = 2304 ----
    int wave = tid >> 5;
    int lane = tid & 31;
    int hh   = lane >> 4;
    int mrow = lane & 15;                    // pixel within row segment
    int nT   = blockIdx.y * 8 + wave;        // out-channel tile 0..15
    int ncol = nT * 16 + (lane & 15);

    v8f acc = {};
    #pragma unroll 2
    for (int kt = 0; kt < CONV_KT; ++kt) {
        int tap = kt >> 3;                   // one 3x3 tap per 32-wide k-block
        int ic0 = (kt & 7) << 5;
        int ky = tap / 3, kx = tap - ky * 3;
        // A fragment: two 16B LDS loads (8 consecutive ic each)
        int abase = (ky * 18 + kx + mrow) * APITCH + ic0 + (hh << 4);
        v8h alo = *(const v8h*)&s_a[abase];
        v8h ahi = *(const v8h*)&s_a[abase + 8];
        v16h a = __builtin_shufflevector(alo, ahi,
                 0,1,2,3,4,5,6,7, 8,9,10,11,12,13,14,15);
        // B fragment: 32B contiguous per lane (pre-swizzled)
        const _Float16* bp = wT + ((((size_t)kt * 16 + nT) * 32) + lane) * 16;
        v16h b;
        #pragma unroll
        for (int e = 0; e < 16; ++e) b[e] = bp[e];
        acc = __builtin_amdgcn_wmma_f32_16x16x32_f16(false, a, false, b, (short)0, acc, false, false);
    }
    float bn = bias[ncol];
    #pragma unroll
    for (int r = 0; r < 8; ++r) {
        int m = r + (hh << 3);
        int xp = x0 + m;
        if (xp < W) {
            float v = acc[r] + bn;
            v = v > 0.0f ? v : 0.0f;                       // relu
            t_out[((size_t)(c_PIXOFF[L] + ty * W + xp)) * 256 + ncol] = (_Float16)v;
        }
    }
}

// ====================================================================
// Kernel 2: 1x1 cls/bbox heads (tiny N -> scalar dot products)
// ====================================================================
__global__ void __launch_bounds__(256)
rpn_heads_kernel(const _Float16* __restrict__ t,
                 const float* __restrict__ cls_w, const float* __restrict__ cls_b,
                 const float* __restrict__ bbox_w, const float* __restrict__ bbox_b,
                 float* __restrict__ logits, float* __restrict__ deltas)
{
    int g = blockIdx.x * 256 + threadIdx.x;
    if (g >= NPIX_TOTAL) return;
    const _Float16* row = t + (size_t)g * 256;
    #pragma unroll 1
    for (int a = 0; a < 3; ++a) {
        const float* wr = cls_w + a * 256;
        float s = cls_b[a];
        for (int c = 0; c < 256; ++c) s += (float)row[c] * wr[c];
        logits[(size_t)g * 3 + a] = s;
    }
    #pragma unroll 1
    for (int cc = 0; cc < 12; ++cc) {
        const float* wr = bbox_w + cc * 256;
        float s = bbox_b[cc];
        for (int c = 0; c < 256; ++c) s += (float)row[c] * wr[c];
        deltas[(size_t)g * 12 + cc] = s;   // (H,W,A,4) order == [p*12 + a*4 + j]
    }
}

// ====================================================================
// Kernel 3: per-level top-k (iterative argmax = sorted) + decode + clip
// ====================================================================
__global__ void __launch_bounds__(256)
topk_decode_kernel(const float* __restrict__ logits, const float* __restrict__ deltas,
                   int* __restrict__ sel_idx, float* __restrict__ sel_score,
                   float* __restrict__ sel_boxes)
{
    __shared__ uint32_t s_used[6000];       // bitmask over up to 192000 candidates
    __shared__ float s_val[256];
    __shared__ int   s_idx[256];
    int L = blockIdx.x;
    int tid = threadIdx.x;
    int base  = c_PIXOFF[L] * 3;
    int ncand = (c_PIXOFF[L+1] - c_PIXOFF[L]) * 3;
    int cnt   = c_CNT[L];
    int nwords = (ncand + 31) >> 5;
    for (int i = tid; i < nwords; i += 256) s_used[i] = 0u;
    __syncthreads();
    const float* lv = logits + base;

    for (int i = 0; i < cnt; ++i) {
        float bv = -3.0e38f; int bi = -1;
        for (int j = tid; j < ncand; j += 256) {
            if (!((s_used[j >> 5] >> (j & 31)) & 1u)) {
                float v = lv[j];
                if (bi < 0 || v > bv || (v == bv && j < bi)) { bv = v; bi = j; }
            }
        }
        s_val[tid] = bv; s_idx[tid] = bi;
        __syncthreads();
        for (int s = 128; s > 0; s >>= 1) {
            if (tid < s) {
                float v2 = s_val[tid + s]; int i2 = s_idx[tid + s];
                float v1 = s_val[tid];     int i1 = s_idx[tid];
                if (i2 >= 0 && (i1 < 0 || v2 > v1 || (v2 == v1 && i2 < i1))) {
                    s_val[tid] = v2; s_idx[tid] = i2;
                }
            }
            __syncthreads();
        }
        if (tid == 0) {
            int b = s_idx[0];
            sel_idx[L * 1000 + i]   = b;
            sel_score[L * 1000 + i] = s_val[0];
            if (b >= 0) s_used[b >> 5] |= (1u << (b & 31));
        }
        __syncthreads();
    }

    int Wl = c_W[L], stride = c_STRIDE[L], sz = c_SIZE[L];
    const float clipv = 4.135166556742356f;  // log(1000/16)
    for (int i = tid; i < cnt; i += 256) {
        int idx = sel_idx[L * 1000 + i];
        int p = idx / 3, a = idx - 3 * p;
        int xp = p - (p / Wl) * Wl, yp = p / Wl;
        float ratio = (a == 0) ? 0.5f : (a == 1) ? 1.0f : 2.0f;
        float wa = sqrtf((float)(sz * sz) / ratio);
        float ha = wa * ratio;
        float shx = (float)(xp * stride), shy = (float)(yp * stride);
        float x1 = shx - 0.5f * wa, y1 = shy - 0.5f * ha;
        float x2 = shx + 0.5f * wa, y2 = shy + 0.5f * ha;
        float bw = x2 - x1 + 1.0f, bh = y2 - y1 + 1.0f;
        float cx = x1 + 0.5f * bw, cy = y1 + 0.5f * bh;
        const float* d = deltas + ((size_t)(c_PIXOFF[L] + p)) * 12 + a * 4;
        float dx = d[0], dy = d[1], dw = d[2], dh = d[3];
        dw = fminf(dw, clipv); dh = fminf(dh, clipv);
        float pcx = dx * bw + cx, pcy = dy * bh + cy;
        float pw = expf(dw) * bw, ph = expf(dh) * bh;
        float bx1 = pcx - 0.5f * pw, by1 = pcy - 0.5f * ph;
        float bx2 = pcx + 0.5f * pw - 1.0f, by2 = pcy + 0.5f * ph - 1.0f;
        bx1 = fminf(fmaxf(bx1, 0.f), 1279.f);
        by1 = fminf(fmaxf(by1, 0.f), 799.f);
        bx2 = fminf(fmaxf(bx2, 0.f), 1279.f);
        by2 = fminf(fmaxf(by2, 0.f), 799.f);
        float* ob = sel_boxes + ((size_t)L * 1000 + i) * 4;
        ob[0] = bx1; ob[1] = by1; ob[2] = bx2; ob[3] = by2;
    }
}

// ====================================================================
// Kernel 4: greedy NMS per level (boxes already score-sorted)
// Box staging uses CDNA5 async global->LDS copy when available.
// ====================================================================
__global__ void __launch_bounds__(256)
nms_kernel(const float* __restrict__ sel_boxes, const float* __restrict__ sel_score,
           float* __restrict__ nms_boxes, float* __restrict__ nms_scores)
{
    __shared__ float s_box[1000 * 4];
    __shared__ unsigned char s_sup[1000];
    __shared__ int s_keep[1000];
    __shared__ int s_head;
    __shared__ int s_chosen;
    int L = blockIdx.x, tid = threadIdx.x;
    int n = c_CNT[L];
    const float* src = sel_boxes + (size_t)L * 4000;
#ifdef HAVE_ASYNC_LDS
    for (int i = tid; i < n * 4; i += 256)
        __builtin_amdgcn_global_load_async_to_lds_b32(
            (__attribute__((address_space(1))) int*)(src + i),
            (__attribute__((address_space(3))) int*)(&s_box[i]), 0, 0);
    __builtin_amdgcn_s_wait_asynccnt(0);
#else
    for (int i = tid; i < n * 4; i += 256) s_box[i] = src[i];
#endif
    for (int i = tid; i < n; i += 256) s_sup[i] = 0;
    if (tid == 0) s_head = 0;
    __syncthreads();

    for (int it = 0; it < 1000; ++it) {
        if (tid == 0) {
            int h = s_head;
            while (h < n && s_sup[h]) ++h;
            s_head = h;
            s_chosen = (h < n) ? h : -1;
            s_keep[it] = s_chosen;
        }
        __syncthreads();
        int ch = s_chosen;
        if (ch >= 0) {
            float bx1 = s_box[ch*4+0], by1 = s_box[ch*4+1];
            float bx2 = s_box[ch*4+2], by2 = s_box[ch*4+3];
            float a1 = (bx2 - bx1 + 1.f) * (by2 - by1 + 1.f);
            for (int j = tid; j < n; j += 256) {
                float x1 = s_box[j*4+0], y1 = s_box[j*4+1];
                float x2 = s_box[j*4+2], y2 = s_box[j*4+3];
                float lx = fmaxf(bx1, x1), ly = fmaxf(by1, y1);
                float rx = fminf(bx2, x2), ry = fminf(by2, y2);
                float iw = fmaxf(rx - lx + 1.f, 0.f), ih = fmaxf(ry - ly + 1.f, 0.f);
                float inter = iw * ih;
                float a2 = (x2 - x1 + 1.f) * (y2 - y1 + 1.f);
                float iou = inter / (a1 + a2 - inter);
                if (iou > 0.7f) s_sup[j] = 1;
            }
        }
        __syncthreads();
    }
    for (int i = tid; i < 1000; i += 256) {
        int k = s_keep[i];
        float* ob = nms_boxes + ((size_t)L * 1000 + i) * 4;
        if (k >= 0) {
            ob[0] = s_box[k*4+0]; ob[1] = s_box[k*4+1];
            ob[2] = s_box[k*4+2]; ob[3] = s_box[k*4+3];
            float lg = sel_score[L * 1000 + k];
            nms_scores[L * 1000 + i] = 1.f / (1.f + expf(-lg));
        } else {
            ob[0] = 0.f; ob[1] = 0.f; ob[2] = 0.f; ob[3] = 0.f;
            nms_scores[L * 1000 + i] = -1.f;
        }
    }
}

// ====================================================================
// Kernel 5: merge 5 levels, top-1000 by score -> proposals
// ====================================================================
__global__ void __launch_bounds__(256)
merge_topk_kernel(const float* __restrict__ nms_boxes, const float* __restrict__ nms_scores,
                  float* __restrict__ props, float* __restrict__ out_props)
{
    __shared__ float s_score[5000];
    __shared__ unsigned char s_used[5000];
    __shared__ float s_val[256];
    __shared__ int   s_idx[256];
    int tid = threadIdx.x;
    for (int i = tid; i < 5000; i += 256) { s_score[i] = nms_scores[i]; s_used[i] = 0; }
    __syncthreads();
    for (int i = 0; i < 1000; ++i) {
        float bv = -3.0e38f; int bi = -1;
        for (int j = tid; j < 5000; j += 256) {
            if (!s_used[j]) {
                float v = s_score[j];
                if (bi < 0 || v > bv || (v == bv && j < bi)) { bv = v; bi = j; }
            }
        }
        s_val[tid] = bv; s_idx[tid] = bi;
        __syncthreads();
        for (int s = 128; s > 0; s >>= 1) {
            if (tid < s) {
                float v2 = s_val[tid + s]; int i2 = s_idx[tid + s];
                float v1 = s_val[tid];     int i1 = s_idx[tid];
                if (i2 >= 0 && (i1 < 0 || v2 > v1 || (v2 == v1 && i2 < i1))) {
                    s_val[tid] = v2; s_idx[tid] = i2;
                }
            }
            __syncthreads();
        }
        if (tid == 0) {
            int b = s_idx[0];
            s_used[b] = 1;
            #pragma unroll
            for (int q = 0; q < 4; ++q) {
                float vq = nms_boxes[(size_t)b * 4 + q];
                props[(size_t)i * 4 + q] = vq;
                out_props[(size_t)i * 4 + q] = vq;
            }
        }
        __syncthreads();
    }
}

// ====================================================================
// Kernel 6: ROI-align (level assignment + masked bilinear + 2x2 avg)
// ====================================================================
__global__ void __launch_bounds__(256)
roi_align_kernel(const float* __restrict__ f0, const float* __restrict__ f1,
                 const float* __restrict__ f2, const float* __restrict__ f3,
                 const float* __restrict__ props, _Float16* __restrict__ xout)
{
    int roi = blockIdx.x;
    int c = threadIdx.x;
    float x1r = props[roi*4+0], y1r = props[roi*4+1];
    float x2r = props[roi*4+2], y2r = props[roi*4+3];
    float area = fmaxf((x2r - x1r) * (y2r - y1r), 1e-6f);
    float lf = floorf(4.0f + log2f(sqrtf(area) / 224.0f + 1e-6f));
    int lvl = (int)(fminf(fmaxf(lf, 2.0f), 5.0f)) - 2;
    const float* feat; int H, W; float scale;
    if      (lvl == 0) { feat = f0; H = 200; W = 320; scale = 0.25f;    }
    else if (lvl == 1) { feat = f1; H = 100; W = 160; scale = 0.125f;   }
    else if (lvl == 2) { feat = f2; H = 50;  W = 80;  scale = 0.0625f;  }
    else               { feat = f3; H = 25;  W = 40;  scale = 0.03125f; }
    float x1 = x1r * scale, y1 = y1r * scale;
    float rw = fmaxf(x2r * scale - x1, 1.0f);
    float rh = fmaxf(y2r * scale - y1, 1.0f);
    const float* fc = feat + (size_t)c * H * W;
    for (int cell = 0; cell < 49; ++cell) {
        int pyc = cell / 7, pxc = cell - (cell / 7) * 7;
        float acc = 0.0f;
        #pragma unroll
        for (int sy = 0; sy < 2; ++sy) {
            #pragma unroll
            for (int sx = 0; sx < 2; ++sx) {
                float gy = ((float)(pyc * 2 + sy) + 0.5f) * 0.5f;  // (j+0.5)/SR
                float gx = ((float)(pxc * 2 + sx) + 0.5f) * 0.5f;
                float yy = y1 + gy * (rh / 7.0f);
                float xx = x1 + gx * (rw / 7.0f);
                bool mask = (yy > -1.f) && (yy < (float)H) && (xx > -1.f) && (xx < (float)W);
                float val = 0.0f;
                if (mask) {
                    float yc = fminf(fmaxf(yy, 0.f), (float)(H - 1));
                    float xc = fminf(fmaxf(xx, 0.f), (float)(W - 1));
                    float y0f = floorf(yc), x0f = floorf(xc);
                    int y0 = (int)y0f, x0 = (int)x0f;
                    int y1i = y0 + 1; if (y1i > H - 1) y1i = H - 1;
                    int x1i = x0 + 1; if (x1i > W - 1) x1i = W - 1;
                    float ly = yc - y0f, lx = xc - x0f;
                    float hy = 1.f - ly, hx = 1.f - lx;
                    val = hy * hx * fc[y0 * W + x0]  + hy * lx * fc[y0 * W + x1i]
                        + ly * hx * fc[y1i * W + x0] + ly * lx * fc[y1i * W + x1i];
                }
                acc += val;
            }
        }
        xout[(size_t)roi * 12544 + (size_t)c * 49 + cell] = (_Float16)(acc * 0.25f);
    }
}

// ====================================================================
// Kernel 7: wave-WMMA GEMM: out = act(A @ W^T + bias)
// A: f16 MxK row-major (K % 32 == 0);  W16: f16 NxK row-major.
// Block = 256 thr (8 waves) -> one 16-row M-tile x 128 N-columns.
// ====================================================================
__global__ void __launch_bounds__(256)
gemm_wmma_kernel(const _Float16* __restrict__ A, int M, int K,
                 const _Float16* __restrict__ Bw, const float* __restrict__ bias,
                 int N, int relu,
                 _Float16* __restrict__ outH, float* __restrict__ outF)
{
    int wave = threadIdx.x >> 5;
    int lane = threadIdx.x & 31;
    int mt = blockIdx.x;
    int nt = blockIdx.y * 8 + wave;
    if (nt * 16 >= N) return;               // whole-wave uniform; no barriers below
    int hh = lane >> 4;
    int m0 = mt * 16, n0 = nt * 16;
    int mrow = m0 + (lane & 15);
    bool mval = mrow < M;
    const _Float16* arow = A + (size_t)(mval ? mrow : 0) * K;
    int ncol = n0 + (lane & 15);
    bool nval = ncol < N;
    const _Float16* wrow = Bw + (size_t)(nval ? ncol : 0) * K;

    v8f acc = {};
    for (int k0 = 0; k0 < K; k0 += 32) {
        __builtin_prefetch(arow + k0 + 128, 0, 0);   // global_prefetch_b8 on A stream
        v16h a, b;
        #pragma unroll
        for (int e = 0; e < 16; ++e) {
            int v = e >> 1, pb = e & 1;
            int k = ((v & 4) << 2) + (hh << 3) + ((v & 3) << 1) + pb;
            a[e] = mval ? arow[k0 + k] : (_Float16)0.0f;
        }
        #pragma unroll
        for (int e = 0; e < 16; ++e)
            b[e] = nval ? wrow[k0 + ((e & 8) << 1) + (hh << 3) + (e & 7)] : (_Float16)0.0f;
        acc = __builtin_amdgcn_wmma_f32_16x16x32_f16(false, a, false, b, (short)0, acc, false, false);
    }
    float bn = nval ? bias[ncol] : 0.0f;
    #pragma unroll
    for (int r = 0; r < 8; ++r) {
        int m = m0 + r + (hh << 3);
        if (m < M && nval) {
            float v = acc[r] + bn;
            if (relu) v = fmaxf(v, 0.0f);
            if (outH) outH[(size_t)m * N + ncol] = (_Float16)v;
            else      outF[(size_t)m * N + ncol] = v;
        }
    }
}

// ====================================================================
// host launcher
// ====================================================================
extern "C" void kernel_launch(void* const* d_in, const int* in_sizes, int n_in,
                              void* d_out, int out_size, void* d_ws, size_t ws_size,
                              hipStream_t stream)
{
    (void)in_sizes; (void)n_in; (void)out_size; (void)ws_size;
    const float* p2 = (const float*)d_in[0];
    const float* p3 = (const float*)d_in[1];
    const float* p4 = (const float*)d_in[2];
    const float* p5 = (const float*)d_in[3];
    const float* p6 = (const float*)d_in[4];
    const float* rpn_conv_w = (const float*)d_in[5];
    const float* rpn_conv_b = (const float*)d_in[6];
    const float* rpn_cls_w  = (const float*)d_in[7];
    const float* rpn_cls_b  = (const float*)d_in[8];
    const float* rpn_bbox_w = (const float*)d_in[9];
    const float* rpn_bbox_b = (const float*)d_in[10];
    const float* fc6_w = (const float*)d_in[11];
    const float* fc6_b = (const float*)d_in[12];
    const float* fc7_w = (const float*)d_in[13];
    const float* fc7_b = (const float*)d_in[14];
    const float* cls_w = (const float*)d_in[15];
    const float* cls_b = (const float*)d_in[16];
    const float* bbox_w = (const float*)d_in[17];
    const float* bbox_b = (const float*)d_in[18];

    char* ws = (char*)d_ws;
    _Float16* t_f16  = (_Float16*)(ws + OFF_T);
    _Float16* x_f16  = (_Float16*)(ws + OFF_X);
    _Float16* h6_f16 = (_Float16*)(ws + OFF_H6);
    _Float16* h7_f16 = (_Float16*)(ws + OFF_H7);
    float* logits    = (float*)(ws + OFF_LG);
    float* deltas    = (float*)(ws + OFF_DL);
    int*   sel_idx   = (int*)  (ws + OFF_SI);
    float* sel_score = (float*)(ws + OFF_SS);
    float* sel_boxes = (float*)(ws + OFF_SB);
    float* nms_boxes = (float*)(ws + OFF_NB);
    float* nms_sc    = (float*)(ws + OFF_NS);
    float* props     = (float*)(ws + OFF_PR);
    _Float16* wT     = (_Float16*)(ws + OFF_WT);
    _Float16* w6_16  = (_Float16*)(ws + OFF_W6);
    _Float16* w7_16  = (_Float16*)(ws + OFF_W7);
    _Float16* wc_16  = (_Float16*)(ws + OFF_WC);
    _Float16* wb_16  = (_Float16*)(ws + OFF_WB);

    float* out_cls   = (float*)d_out;                 // [1000][81]
    float* out_bbox  = out_cls + 1000 * 81;           // [1000][324]
    float* out_props = out_cls + 1000 * (81 + 324);   // [1000][4]

    // 0) weight prep: swizzle conv weights, convert fc weights to f16
    swizzle_convw_kernel<<<dim3((CONV_KT*16*32 + 255)/256), 256, 0, stream>>>(rpn_conv_w, wT);
    cvt_f16_kernel<<<dim3((1024*12544 + 255)/256), 256, 0, stream>>>(fc6_w, w6_16, 1024*12544);
    cvt_f16_kernel<<<dim3((1024*1024  + 255)/256), 256, 0, stream>>>(fc7_w, w7_16, 1024*1024);
    cvt_f16_kernel<<<dim3((81*1024    + 255)/256), 256, 0, stream>>>(cls_w, wc_16, 81*1024);
    cvt_f16_kernel<<<dim3((324*1024   + 255)/256), 256, 0, stream>>>(bbox_w, wb_16, 324*1024);

    // 1) RPN 3x3 conv + relu -> t (f16); LDS-staged implicit GEMM
    rpn_conv3x3_kernel<<<dim3(NTILES_TOTAL, 2), 256, 0, stream>>>(
        p2, p3, p4, p5, p6, wT, rpn_conv_b, t_f16);
    // 2) 1x1 cls/bbox heads
    rpn_heads_kernel<<<dim3((NPIX_TOTAL + 255) / 256), 256, 0, stream>>>(
        t_f16, rpn_cls_w, rpn_cls_b, rpn_bbox_w, rpn_bbox_b, logits, deltas);
    // 3) per-level top-k + decode
    topk_decode_kernel<<<5, 256, 0, stream>>>(logits, deltas, sel_idx, sel_score, sel_boxes);
    // 4) per-level NMS
    nms_kernel<<<5, 256, 0, stream>>>(sel_boxes, sel_score, nms_boxes, nms_sc);
    // 5) merge -> proposals (ws + d_out tail)
    merge_topk_kernel<<<1, 256, 0, stream>>>(nms_boxes, nms_sc, props, out_props);
    // 6) ROI align -> pooled f16 [1000][12544]
    roi_align_kernel<<<1000, 256, 0, stream>>>(p2, p3, p4, p5, props, x_f16);
    // 7) box head GEMMs (WMMA, 8-wave blocks)
    gemm_wmma_kernel<<<dim3(63, 8), 256, 0, stream>>>(x_f16, 1000, 12544, w6_16, fc6_b, 1024, 1, h6_f16, nullptr);
    gemm_wmma_kernel<<<dim3(63, 8), 256, 0, stream>>>(h6_f16, 1000, 1024, w7_16, fc7_b, 1024, 1, h7_f16, nullptr);
    gemm_wmma_kernel<<<dim3(63, 1), 256, 0, stream>>>(h7_f16, 1000, 1024, wc_16, cls_b, 81,  0, nullptr, out_cls);
    gemm_wmma_kernel<<<dim3(63, 3), 256, 0, stream>>>(h7_f16, 1000, 1024, wb_16, bbox_b, 324, 0, nullptr, out_bbox);
}
